// MultiHeadAttention_60275571032298
// MI455X (gfx1250) — matrix operand order repaired and checked by reading, not compiled, not run on hardware
//
#include <hip/hip_runtime.h>
#include <hip/hip_bf16.h>

typedef __bf16 bf16_t;
typedef __attribute__((ext_vector_type(16))) __bf16 v16bf;
typedef __attribute__((ext_vector_type(8)))  __bf16 v8bf;
typedef __attribute__((ext_vector_type(4)))  __bf16 v4bf;
typedef __attribute__((ext_vector_type(8)))  float  v8f;
typedef __attribute__((ext_vector_type(4)))  unsigned int v4u;
typedef __attribute__((ext_vector_type(8)))  unsigned int v8u;

#define HIDDEN 512
#define HEADS  8
#define HD     64
#define SEQ    2048

union BF16x16 { v16bf v; v8bf h[2]; };

// ---- WMMA fragment loaders (CDNA5 16x16x32 bf16 layouts, wave32) ----

// A-matrix 16x32 (MxK): lanes 0-15 = rows; per lane two 16B chunks at
// K = 8*g..8*g+7 and K = 8*g+16..8*g+23 (g = lane>>4).
__device__ __forceinline__ v16bf load_frag_a(const bf16_t* base, int ld) {
  int lane = threadIdx.x & 31;
  const bf16_t* p = base + (size_t)(lane & 15) * ld + ((lane >> 4) << 3);
  BF16x16 f;
  f.h[0] = *(const v8bf*)(p);
  f.h[1] = *(const v8bf*)(p + 16);
  return f.v;
}

// B-matrix 32x16 (KxN): lane holds column n = lane&15, K = 8*g..8*g+7 and 8*g+16..8*g+23 (g = lane>>4).
// Source stored [n][k] row-major (k contiguous) with row stride ld.
__device__ __forceinline__ v16bf load_frag_b(const bf16_t* base, int ld) {
  int lane = threadIdx.x & 31;
  const bf16_t* p = base + (size_t)(lane & 15) * ld + ((lane >> 4) << 3);
  BF16x16 f;
  f.h[0] = *(const v8bf*)(p);
  f.h[1] = *(const v8bf*)(p + 16);
  return f.v;
}

__device__ __forceinline__ v8f wmma_bf16(v16bf a, v16bf b, v8f c) {
  return __builtin_amdgcn_wmma_f32_16x16x32_bf16(false, a, false, b,
                                                 (short)0, c, false, false);
}

__device__ __forceinline__ v8f v8f_zero() {
  v8f z = {0.f, 0.f, 0.f, 0.f, 0.f, 0.f, 0.f, 0.f};
  return z;
}

// XOR butterfly exchange via ds_swizzle_b32 (group-of-32 mode):
// new_lane = (lane & 0x1f) ^ MASK, pattern is an immediate -> no index VALU.
template<int MASK>
__device__ __forceinline__ float swz_xor(float x) {
  int i = __float_as_int(x);
  i = __builtin_amdgcn_ds_swizzle(i, (MASK << 10) | 0x1f);
  return __int_as_float(i);
}

// ---- TDM: async-load a 2D bf16 tile (tile1 rows x tile0 elems) into LDS ----
// D# per CDNA5 ISA; 2D -> 2 SGPR groups.
// Rows packed contiguously in LDS (tile0*2 bytes apart).
__device__ __forceinline__ void tdm_load_2d(const bf16_t* gsrc, bf16_t* ldst,
                                            unsigned tile0, unsigned tile1,
                                            unsigned stride0, unsigned tdim1) {
  unsigned long long ga = (unsigned long long)(uintptr_t)gsrc;
  unsigned la = (unsigned)(uintptr_t)ldst;        // low 32b = LDS offset
  v4u g0;
  g0[0] = 1u;                                     // count=1, user mode
  g0[1] = la;                                     // lds_addr (bytes)
  g0[2] = (unsigned)ga;                           // global_addr[31:0]
  g0[3] = (unsigned)(ga >> 32) | (2u << 30);      // addr[56:32] | type=2
  v8u g1;
  g1[0] = (1u << 16);                             // wg_mask=0, data_size=2B
  g1[1] = (stride0 & 0xFFFFu) << 16;              // tensor_dim0 (low half)
  g1[2] = (stride0 >> 16) | ((tdim1 & 0xFFFFu) << 16);  // dim0 hi | dim1 lo
  g1[3] = (tdim1 >> 16) | (tile0 << 16);          // dim1 hi | tile_dim0
  g1[4] = tile1;                                  // tile_dim1, tile_dim2=0
  g1[5] = stride0;                                // tensor_dim0_stride low 32
  g1[6] = 0u;
  g1[7] = 0u;
  asm volatile("tensor_load_to_lds %0, %1" :: "s"(g0), "s"(g1) : "memory");
}

// ---- f32 -> bf16 conversion, 4 elements/thread ----
__global__ void cvt_f32_bf16(const float* __restrict__ src,
                             bf16_t* __restrict__ dst, int n4) {
  int i = blockIdx.x * blockDim.x + threadIdx.x;
  if (i >= n4) return;
  float4 f = ((const float4*)src)[i];
  v4bf o;
  o[0] = (bf16_t)f.x; o[1] = (bf16_t)f.y;
  o[2] = (bf16_t)f.z; o[3] = (bf16_t)f.w;
  ((v4bf*)dst)[i] = o;
}

// ---- GEMM: out[m,n] = (sum_k A[m,k]*W[n,k] + bias[n]) * scale ----
// MODE 0: f32 row-major [M,N]
// MODE 1: bf16 [B,H,S,D]   (Q / K)
// MODE 2: bf16 [B,H,D,S]   (V transposed for PV B-fragments)
// B-panel staged through LDS by the Tensor Data Mover, double-buffered.
template<int MODE>
__global__ __launch_bounds__(128) void gemm_wmma(
    const bf16_t* __restrict__ A, const bf16_t* __restrict__ W,
    const float* __restrict__ bias, void* __restrict__ out,
    int M, float scale)
{
  constexpr int N = HIDDEN, K = HIDDEN;
  constexpr int KSTEPS = K / 32;
  __shared__ bf16_t bpanel[2][128 * 32];   // 2 x 8 KB

  int lane = threadIdx.x & 31;
  int wid  = threadIdx.x >> 5;
  int m0  = blockIdx.x * 128 + (wid >> 1) * 64;
  int n0w = (wid & 1) * 64;                  // wave's n-offset inside panel
  int n0  = blockIdx.y * 128 + n0w;
  const bf16_t* wpanel = W + (size_t)(blockIdx.y * 128) * K;

  v8f acc[4][4];
#pragma unroll
  for (int i = 0; i < 4; i++)
#pragma unroll
    for (int j = 0; j < 4; j++) acc[i][j] = v8f_zero();

  // prologue: TDM-load panel for k-step 0
  if (wid == 0) {
    tdm_load_2d(wpanel, &bpanel[0][0], 32, 128, K, N);
    __builtin_amdgcn_s_wait_tensorcnt(0);
  }
  __syncthreads();

  for (int ks = 0; ks < KSTEPS; ks++) {
    int cur = ks & 1;
    if (wid == 0 && ks + 1 < KSTEPS)         // overlap next TDM with compute
      tdm_load_2d(wpanel + (ks + 1) * 32, &bpanel[cur ^ 1][0], 32, 128, K, N);

    int k0 = ks * 32;
    // per-lane prefetch of the A panel two k-steps ahead (64 rows, 2/lane)
    if (ks + 2 < KSTEPS) {
      __builtin_prefetch(A + (size_t)(m0 + lane * 2) * K + k0 + 64, 0, 1);
      __builtin_prefetch(A + (size_t)(m0 + lane * 2 + 1) * K + k0 + 64, 0, 1);
    }

    v16bf a[4], b[4];
#pragma unroll
    for (int i = 0; i < 4; i++)
      a[i] = load_frag_a(A + (size_t)(m0 + i * 16) * K + k0, K);
#pragma unroll
    for (int j = 0; j < 4; j++)
      b[j] = load_frag_b(&bpanel[cur][(n0w + j * 16) * 32], 32);
#pragma unroll
    for (int i = 0; i < 4; i++)
#pragma unroll
      for (int j = 0; j < 4; j++)
        acc[i][j] = wmma_bf16(a[i], b[j], acc[i][j]);

    if (wid == 0 && ks + 1 < KSTEPS)
      __builtin_amdgcn_s_wait_tensorcnt(0);
    __syncthreads();
  }

  // Epilogue. C-layout: VGPR r, group g = lane>>4: row = r + 8*g, col = lane&15.
  int g = lane >> 4, c = lane & 15;
#pragma unroll
  for (int j = 0; j < 4; j++) {
    float bn = bias[n0 + j * 16 + c];
#pragma unroll
    for (int i = 0; i < 4; i++) {
#pragma unroll
      for (int r = 0; r < 8; r++) {
        float v = (acc[i][j][r] + bn) * scale;
        int row = m0 + i * 16 + r + g * 8;
        int col = n0 + j * 16 + c;
        if constexpr (MODE == 0) {
          ((float*)out)[(size_t)row * N + col] = v;
        } else {
          int b_ = row >> 11, s_ = row & (SEQ - 1);   // SEQ = 2048
          int h_ = col >> 6,  d_ = col & 63;
          size_t dst;
          if constexpr (MODE == 1)
            dst = ((size_t)(b_ * HEADS + h_) * SEQ + s_) * HD + d_;
          else
            dst = ((size_t)(b_ * HEADS + h_) * HD + d_) * SEQ + s_;
          ((bf16_t*)out)[dst] = (bf16_t)v;
        }
      }
    }
  }
}

// ---- Flash attention ----
// Workgroup = 4 waves x 32 query rows = 128 rows; K/V panels (32 keys/step)
// are TDM-staged into LDS once per workgroup and shared by all 4 waves.
// Q: [B,H,S,D] bf16 (pre-scaled by 1/sqrt(D)); K: [B,H,S,D]; Vt: [B,H,D,S].
__global__ __launch_bounds__(128) void attn_wmma(
    const bf16_t* __restrict__ Q, const bf16_t* __restrict__ Kc,
    const bf16_t* __restrict__ Vt, bf16_t* __restrict__ ctx)
{
  constexpr int S = SEQ;
  __shared__ bf16_t ldsK[2][32 * HD];    // 2 x 4 KB: [key][d], d contiguous
  __shared__ bf16_t ldsV[2][HD * 32];    // 2 x 4 KB: [d][key], key contiguous
  __shared__ bf16_t lds_p[4][16 * 32];   // per-wave P tile (C->A relayout)

  int lane = threadIdx.x & 31;
  int wid  = threadIdx.x >> 5;
  int bh = blockIdx.y;
  int b_ = bh >> 3, h_ = bh & 7;
  const bf16_t* Qb = Q  + (size_t)bh * S * HD;
  const bf16_t* Kb = Kc + (size_t)bh * S * HD;
  const bf16_t* Vb = Vt + (size_t)bh * HD * S;
  int m0 = blockIdx.x * 128 + wid * 32;  // 2 m-tiles of 16 rows per wave

  v16bf qa[2][2];
#pragma unroll
  for (int mt = 0; mt < 2; mt++) {
    qa[mt][0] = load_frag_a(Qb + (size_t)(m0 + mt * 16) * HD, HD);
    qa[mt][1] = load_frag_a(Qb + (size_t)(m0 + mt * 16) * HD + 32, HD);
  }

  v8f o[2][4];
  float mrow[2][8], lrow[2][8];
#pragma unroll
  for (int mt = 0; mt < 2; mt++) {
#pragma unroll
    for (int t = 0; t < 4; t++) o[mt][t] = v8f_zero();
#pragma unroll
    for (int r = 0; r < 8; r++) { mrow[mt][r] = -1e30f; lrow[mt][r] = 0.f; }
  }

  int g = lane >> 4, c = lane & 15;
  bf16_t* lp = lds_p[wid];

  // prologue: TDM-load K/V panels for j = 0
  if (wid == 0) {
    tdm_load_2d(Kb, &ldsK[0][0], HD, 32, HD, S);      // 32 keys x 64 d
    tdm_load_2d(Vb, &ldsV[0][0], 32, HD, S, HD);      // 64 d x 32 keys
    __builtin_amdgcn_s_wait_tensorcnt(0);
  }
  __syncthreads();

  for (int j = 0; j < S; j += 32) {
    int cur = (j >> 5) & 1;
    if (wid == 0 && j + 32 < S) {                     // overlap next TDM
      tdm_load_2d(Kb + (size_t)(j + 32) * HD, &ldsK[cur ^ 1][0], HD, 32, HD, S);
      tdm_load_2d(Vb + (j + 32), &ldsV[cur ^ 1][0], 32, HD, S, HD);
    }
    const bf16_t* kk = &ldsK[cur][0];
    const bf16_t* vv = &ldsV[cur][0];

#pragma unroll
    for (int mt = 0; mt < 2; mt++) {
      // scores: two 16-key tiles, accumulated over d = 0..63 (from LDS)
      v8f s0 = v8f_zero(), s1 = v8f_zero();
      s0 = wmma_bf16(qa[mt][0], load_frag_b(kk, HD), s0);
      s0 = wmma_bf16(qa[mt][1], load_frag_b(kk + 32, HD), s0);
      s1 = wmma_bf16(qa[mt][0], load_frag_b(kk + 16 * HD, HD), s1);
      s1 = wmma_bf16(qa[mt][1], load_frag_b(kk + 16 * HD + 32, HD), s1);

      // online softmax: row stats per C-layout slot (row = r + 8*g);
      // 16-lane butterfly reductions via immediate-pattern ds_swizzle.
      float alpha[8];
#pragma unroll
      for (int r = 0; r < 8; r++) {
        float mx = fmaxf(s0[r], s1[r]);
        mx = fmaxf(mx, swz_xor<1>(mx));
        mx = fmaxf(mx, swz_xor<2>(mx));
        mx = fmaxf(mx, swz_xor<4>(mx));
        mx = fmaxf(mx, swz_xor<8>(mx));
        float mnew = fmaxf(mrow[mt][r], mx);
        alpha[r] = __expf(mrow[mt][r] - mnew);
        float p0 = __expf(s0[r] - mnew);
        float p1 = __expf(s1[r] - mnew);
        s0[r] = p0; s1[r] = p1;
        float rs = p0 + p1;
        rs += swz_xor<1>(rs);
        rs += swz_xor<2>(rs);
        rs += swz_xor<4>(rs);
        rs += swz_xor<8>(rs);
        lrow[mt][r] = lrow[mt][r] * alpha[r] + rs;
        mrow[mt][r] = mnew;
      }
#pragma unroll
      for (int t = 0; t < 4; t++)
#pragma unroll
        for (int r = 0; r < 8; r++) o[mt][t][r] *= alpha[r];

      // P: C-layout -> row-major 16x32 bf16 in LDS -> reload as A-fragment.
      // Same-wave DS ops are in-order; compiler inserts dscnt waits.
#pragma unroll
      for (int r = 0; r < 8; r++) {
        lp[(r + 8 * g) * 32 + c]      = (bf16_t)s0[r];
        lp[(r + 8 * g) * 32 + 16 + c] = (bf16_t)s1[r];
      }
      v16bf pa = load_frag_a(lp, 32);

      // O += P(16x32) x V(32x16) for the 4 d-tiles (LDS, keys contiguous)
#pragma unroll
      for (int t = 0; t < 4; t++)
        o[mt][t] = wmma_bf16(pa, load_frag_b(vv + (t * 16) * 32, 32), o[mt][t]);
    }

    if (wid == 0 && j + 32 < S)
      __builtin_amdgcn_s_wait_tensorcnt(0);
    __syncthreads();
  }

  // normalize and store ctx [B,S,HIDDEN]
  bf16_t* cb = ctx + (size_t)b_ * S * HIDDEN;
#pragma unroll
  for (int mt = 0; mt < 2; mt++) {
#pragma unroll
    for (int r = 0; r < 8; r++) lrow[mt][r] = 1.f / lrow[mt][r];
#pragma unroll
    for (int t = 0; t < 4; t++)
#pragma unroll
      for (int r = 0; r < 8; r++) {
        int row = m0 + mt * 16 + r + 8 * g;
        int col = h_ * HD + t * 16 + c;
        cb[(size_t)row * HIDDEN + col] = (bf16_t)(o[mt][t][r] * lrow[mt][r]);
      }
  }
}

extern "C" void kernel_launch(void* const* d_in, const int* in_sizes, int n_in,
                              void* d_out, int out_size, void* d_ws, size_t ws_size,
                              hipStream_t stream) {
  const float* x  = (const float*)d_in[0];
  const float* Wq = (const float*)d_in[1];
  const float* bq = (const float*)d_in[2];
  const float* Wk = (const float*)d_in[3];
  const float* bk = (const float*)d_in[4];
  const float* Wv = (const float*)d_in[5];
  const float* bv = (const float*)d_in[6];
  const float* Wo = (const float*)d_in[7];
  const float* bo = (const float*)d_in[8];

  const int M = in_sizes[0] / HIDDEN;   // B*S
  const int B = M / SEQ;

  char* ws = (char*)d_ws;
  size_t off = 0;
  auto alloc = [&](size_t bytes) -> char* {
    char* p = ws + off;
    off += (bytes + 255) & ~(size_t)255;
    return p;
  };
  bf16_t* xb   = (bf16_t*)alloc((size_t)M * HIDDEN * sizeof(bf16_t));
  bf16_t* wqb  = (bf16_t*)alloc((size_t)HIDDEN * HIDDEN * sizeof(bf16_t));
  bf16_t* wkb  = (bf16_t*)alloc((size_t)HIDDEN * HIDDEN * sizeof(bf16_t));
  bf16_t* wvb  = (bf16_t*)alloc((size_t)HIDDEN * HIDDEN * sizeof(bf16_t));
  bf16_t* wob  = (bf16_t*)alloc((size_t)HIDDEN * HIDDEN * sizeof(bf16_t));
  bf16_t* Qb   = (bf16_t*)alloc((size_t)M * HIDDEN * sizeof(bf16_t));
  bf16_t* Kb   = (bf16_t*)alloc((size_t)M * HIDDEN * sizeof(bf16_t));
  bf16_t* Vtb  = (bf16_t*)alloc((size_t)M * HIDDEN * sizeof(bf16_t));
  bf16_t* ctxb = (bf16_t*)alloc((size_t)M * HIDDEN * sizeof(bf16_t));

  int nx4 = (M * HIDDEN) / 4;
  cvt_f32_bf16<<<(nx4 + 255) / 256, 256, 0, stream>>>(x, xb, nx4);
  int nw4 = (HIDDEN * HIDDEN) / 4;
  cvt_f32_bf16<<<(nw4 + 255) / 256, 256, 0, stream>>>(Wq, wqb, nw4);
  cvt_f32_bf16<<<(nw4 + 255) / 256, 256, 0, stream>>>(Wk, wkb, nw4);
  cvt_f32_bf16<<<(nw4 + 255) / 256, 256, 0, stream>>>(Wv, wvb, nw4);
  cvt_f32_bf16<<<(nw4 + 255) / 256, 256, 0, stream>>>(Wo, wob, nw4);

  dim3 gg(M / 128, HIDDEN / 128), bb(128);
  gemm_wmma<1><<<gg, bb, 0, stream>>>(xb, wqb, bq, Qb,  M, 0.125f);
  gemm_wmma<1><<<gg, bb, 0, stream>>>(xb, wkb, bk, Kb,  M, 1.0f);
  gemm_wmma<2><<<gg, bb, 0, stream>>>(xb, wvb, bv, Vtb, M, 1.0f);

  attn_wmma<<<dim3(SEQ / 128, B * HEADS), 128, 0, stream>>>(Qb, Kb, Vtb, ctxb);

  gemm_wmma<0><<<gg, bb, 0, stream>>>(ctxb, wob, bo, d_out, M, 1.0f);
}
